// SinkhornKnopp_6640019440346
// MI455X (gfx1250) — matrix operand order repaired and checked by reading, not compiled, not run on hardware
//
#include <hip/hip_runtime.h>
#include <hip/hip_bf16.h>
#include <stdint.h>

#define SK_EPS 1e-8f

typedef __attribute__((ext_vector_type(16))) __bf16 v16bf;
typedef __attribute__((ext_vector_type(8)))  __bf16 v8bf;
typedef __attribute__((ext_vector_type(8)))  float  v8f;
typedef __attribute__((ext_vector_type(4)))  unsigned int u32x4;
typedef __attribute__((ext_vector_type(8)))  int i32x8;
typedef __attribute__((ext_vector_type(4)))  int i32x4;

#ifndef __has_builtin
#define __has_builtin(x) 0
#endif
#if __has_builtin(__builtin_amdgcn_tensor_load_to_lds) && __has_builtin(__builtin_amdgcn_s_wait_tensorcnt)
#define SK_HAVE_TDM 1
#else
#define SK_HAVE_TDM 0
#endif

__device__ __forceinline__ unsigned short sk_f32_bf16(float f) {
  unsigned int x = __float_as_uint(f);
  x += 0x7FFFu + ((x >> 16) & 1u);           // round-to-nearest-even
  return (unsigned short)(x >> 16);
}
__device__ __forceinline__ float sk_bf16_f32(unsigned int h) {
  return __uint_as_float(h << 16);
}

// ---------------------------------------------------------------------------
// K = exp(H) in bf16, written into the first 128MB of d_out. 8 elems/thread.
// ---------------------------------------------------------------------------
__global__ void sk_exp_bf16(const float* __restrict__ H, unsigned short* __restrict__ K) {
  size_t e0 = ((size_t)blockIdx.x * 256 + threadIdx.x) * 8;
  float4 h0 = *(const float4*)(H + e0);
  float4 h1 = *(const float4*)(H + e0 + 4);
  unsigned int p0 = (unsigned int)sk_f32_bf16(expf(h0.x)) | ((unsigned int)sk_f32_bf16(expf(h0.y)) << 16);
  unsigned int p1 = (unsigned int)sk_f32_bf16(expf(h0.z)) | ((unsigned int)sk_f32_bf16(expf(h0.w)) << 16);
  unsigned int p2 = (unsigned int)sk_f32_bf16(expf(h1.x)) | ((unsigned int)sk_f32_bf16(expf(h1.y)) << 16);
  unsigned int p3 = (unsigned int)sk_f32_bf16(expf(h1.z)) | ((unsigned int)sk_f32_bf16(expf(h1.w)) << 16);
  *(uint4*)(K + e0) = make_uint4(p0, p1, p2, p3);
}

// u,v stored contiguously in d_ws: 2 * 64 * 1024 floats -> init to 1.
__global__ void sk_init_uv(float* __restrict__ uv) {
  uv[(size_t)blockIdx.x * 256 + threadIdx.x] = 1.0f;
}

// ---------------------------------------------------------------------------
// Col phase, part 1: partial t_j = sum_{i in chunk} u_i * K[i][j].
// grid (4 row-chunks, 64 batches), 256 threads, 4 cols/thread (8B bf16 loads,
// fully coalesced). u_i is block-uniform -> scalar loads.
// ---------------------------------------------------------------------------
__global__ void sk_col_partial(const unsigned short* __restrict__ K,
                               const float* __restrict__ u,
                               float* __restrict__ tp) {
  const int b = blockIdx.y, ch = blockIdx.x;
  const int col = threadIdx.x * 4;
  const unsigned short* kp = K + ((size_t)b << 20) + ((size_t)(ch << 8) << 10) + col;
  const float* up = u + (b << 10) + (ch << 8);
  float a0 = 0.f, a1 = 0.f, a2 = 0.f, a3 = 0.f;
#pragma unroll 4
  for (int i = 0; i < 256; ++i) {
    float ui = up[i];
    uint2 pk = *(const uint2*)(kp + (size_t)i * 1024);
    a0 = fmaf(ui, sk_bf16_f32(pk.x & 0xFFFFu), a0);
    a1 = fmaf(ui, sk_bf16_f32(pk.x >> 16),     a1);
    a2 = fmaf(ui, sk_bf16_f32(pk.y & 0xFFFFu), a2);
    a3 = fmaf(ui, sk_bf16_f32(pk.y >> 16),     a3);
  }
  *(float4*)(tp + ((((size_t)(b << 2) + ch) << 10) + col)) = make_float4(a0, a1, a2, a3);
}

// Col phase, part 2: combine 4 partials, v_j = v_j / (v_j * t_j + eps)
__global__ void sk_col_combine(const float* __restrict__ tp, float* __restrict__ v) {
  int gid = blockIdx.x * 256 + threadIdx.x;   // b*1024 + j
  int b = gid >> 10, j = gid & 1023;
  size_t base = ((size_t)b << 12) + j;
  float t = tp[base] + tp[base + 1024] + tp[base + 2048] + tp[base + 3072];
  float vo = v[gid];
  v[gid] = vo / fmaf(vo, t, SK_EPS);
}

// ---------------------------------------------------------------------------
// Row phase: s_i = sum_j K[i][j]*v_j via WMMA bf16, then u update.
// Block = 128 threads (4 waves x 16 rows), tile = 64 rows x 1024 cols bf16.
// Tile brought into LDS by TDM with 1-dword-per-256-dword padding so that the
// 16-row-strided WMMA A reads are LDS bank-conflict free.
// Padded dword address of linear dword l: p = l + (l >> 8).
// ---------------------------------------------------------------------------
__global__ __launch_bounds__(128) void sk_row_wmma(const unsigned short* __restrict__ K,
                                                   const float* __restrict__ v,
                                                   float* __restrict__ u) {
  __shared__ __align__(16) unsigned int   tile[32896];   // 64*512 dw + 128 pad dw = 128.5KB
  __shared__ __align__(16) unsigned short vls[1024];     // v in bf16
  __shared__ float slds[64];

  const int b  = blockIdx.y;
  const int i0 = blockIdx.x << 6;                        // 64 rows per block

  // stage v (bf16) into LDS
  for (int idx = threadIdx.x; idx < 1024; idx += 128)
    vls[idx] = sk_f32_bf16(v[(b << 10) + idx]);

  const unsigned short* gsrc = K + ((size_t)b << 20) + ((size_t)i0 << 10);

#if SK_HAVE_TDM
  if (threadIdx.x < 32) {                                // one wave issues the DMA
    unsigned long long ga = (unsigned long long)(uintptr_t)gsrc;
    u32x4 g0 = { 1u,                                     // count=1 user descriptor
                 (unsigned int)(uintptr_t)(void*)tile,   // lds_addr
                 (unsigned int)(ga & 0xFFFFFFFFu),
                 (unsigned int)((ga >> 32) & 0x01FFFFFFu) | 0x80000000u }; // addr[56:32] | type=2
    i32x8 g1 = { (int)(0x00010000u | (1u << 20) | (7u << 22)), // data_size=2B, pad_en, 256dw interval, 1dw pad
                 (int)(1024u << 16),                     // tensor_dim0 = 1024
                 (int)(64u << 16),                       // tensor_dim1 = 64
                 (int)(1024u << 16),                     // tile_dim0 = 1024
                 64,                                     // tile_dim1 = 64
                 1024,                                   // tensor_dim0_stride = 1024
                 0, 0 };
    i32x4 gz = { 0, 0, 0, 0 };
#if defined(__clang_major__) && (__clang_major__ >= 23)
    i32x8 gz8 = { 0, 0, 0, 0, 0, 0, 0, 0 };
    __builtin_amdgcn_tensor_load_to_lds(g0, g1, gz, gz, gz8, 0);
#else
    __builtin_amdgcn_tensor_load_to_lds(g0, g1, gz, gz, 0);
#endif
    __builtin_amdgcn_s_wait_tensorcnt(0);
  }
#else
  {  // fallback: cooperative coalesced copy with the same padded layout
    const unsigned int* gdw = (const unsigned int*)gsrc;
    for (int l = threadIdx.x; l < 32768; l += 128) tile[l + (l >> 8)] = gdw[l];
  }
#endif
  __syncthreads();

  const int lane = threadIdx.x & 31;
  const int w    = threadIdx.x >> 5;
  const int h    = lane >> 4;                  // lane half selects K sub-chunk
  const int r    = (w << 4) + (lane & 15);     // local row 0..63

  v8f c = {0.f, 0.f, 0.f, 0.f, 0.f, 0.f, 0.f, 0.f};
  const v8bf* vv = (const v8bf*)vls;

  for (int j0 = 0; j0 < 1024; j0 += 32) {
    // A (16x32 bf16): lane needs cols [j0+8h, +8) and [j0+16+8h, +8) of its row
    int l1 = (r << 9) + (j0 >> 1) + (h << 2);
    int p1 = l1 + (l1 >> 8);
    int l2 = l1 + 8;
    int p2 = l2 + (l2 >> 8);
    v8bf a0 = *(const v8bf*)(tile + p1);
    v8bf a1 = *(const v8bf*)(tile + p2);
    v16bf A = __builtin_shufflevector(a0, a1, 0,1,2,3,4,5,6,7,8,9,10,11,12,13,14,15);
    // B (32x16 bf16): B[k][n] = v[j0+k] for all n -> lane half holds v[j0+8h .. +7] and v[j0+16+8h .. +7]
    int q = (j0 >> 3) + h;
    v8bf b0 = vv[q];
    v8bf b1 = vv[q + 2];
    v16bf B = __builtin_shufflevector(b0, b1, 0,1,2,3,4,5,6,7,8,9,10,11,12,13,14,15);
    c = __builtin_amdgcn_wmma_f32_16x16x32_bf16(false, A, false, B, (short)0, c, false, false);
  }

  // D[m][n] independent of n: lane 0 of each half holds s for rows 16w+8h+p in c[p]
  if ((lane & 15) == 0) {
#pragma unroll
    for (int p = 0; p < 8; ++p) slds[(w << 4) + (h << 3) + p] = c[p];
  }
  __syncthreads();

  if (threadIdx.x < 64) {
    float s = slds[threadIdx.x];
    size_t gi = ((size_t)b << 10) + i0 + threadIdx.x;
    float uo = u[gi];
    u[gi] = uo / fmaf(uo, s, SK_EPS);
  }
}

// ---------------------------------------------------------------------------
// Final: M = exp(H) * u_i * v_j in fp32 (recompute exp from H for full
// precision; overwrites the bf16 K scratch living in d_out).
// ---------------------------------------------------------------------------
__global__ void sk_final(const float* __restrict__ H, const float* __restrict__ u,
                         const float* __restrict__ v, float* __restrict__ out) {
  size_t e0 = ((size_t)blockIdx.x * 256 + threadIdx.x) * 4;
  size_t row = e0 >> 10;                      // b*1024 + i
  int b = (int)(row >> 10);
  int jc = (int)(e0 & 1023);
  float ui = u[row];
  float4 vj = *(const float4*)(v + ((size_t)b << 10) + jc);
  float4 hh = *(const float4*)(H + e0);
  float4 o;
  o.x = expf(hh.x) * ui * vj.x;
  o.y = expf(hh.y) * ui * vj.y;
  o.z = expf(hh.z) * ui * vj.z;
  o.w = expf(hh.w) * ui * vj.w;
  *(float4*)(out + e0) = o;
}

extern "C" void kernel_launch(void* const* d_in, const int* in_sizes, int n_in,
                              void* d_out, int out_size, void* d_ws, size_t ws_size,
                              hipStream_t stream) {
  const float* H = (const float*)d_in[0];
  unsigned short* K = (unsigned short*)d_out;       // bf16 K scratch (first 128MB of d_out)
  float* out = (float*)d_out;
  float* u  = (float*)d_ws;                         // 64*1024 floats
  float* v  = u + 65536;                            // 64*1024 floats
  float* tp = v + 65536;                            // 64*4*1024 floats (col partials)

  sk_exp_bf16<<<32768, 256, 0, stream>>>(H, K);     // 67108864 / 8 / 256
  sk_init_uv<<<512, 256, 0, stream>>>(u);           // inits u and v (contiguous)

  for (int it = 0; it < 20; ++it) {
    sk_col_partial<<<dim3(4, 64), 256, 0, stream>>>(K, u, tp);
    sk_col_combine<<<256, 256, 0, stream>>>(tp, v);
    sk_row_wmma<<<dim3(16, 64), 128, 0, stream>>>(K, v, u);
  }

  sk_final<<<65536, 256, 0, stream>>>(H, u, v, out); // 67108864 / 4 / 256
}
